// SwinTransformerBlock_85263690760284
// MI455X (gfx1250) — matrix operand order repaired and checked by reading, not compiled, not run on hardware
//
#include <hip/hip_runtime.h>
#include <hip/hip_bf16.h>

typedef _Float16 half_t;
typedef __attribute__((ext_vector_type(16))) _Float16 v16h;
typedef __attribute__((ext_vector_type(8)))  float    v8f;
typedef unsigned int u32;
typedef __attribute__((ext_vector_type(4))) u32 v4u;

#define TOK     100352u      // B*NW*N = 32*64*49
#define CDIM    384
#define HIDDIM  1536
#define NWIN    2048
#define NHEADS  12
#define NPAD    64
#define HD      32
#define SCALE_Q 0.17677669529663687f

union HV { struct { v4u lo, hi; } u; v16h h; };

// ---- fragment loaders (CDNA5 WMMA 16x16x32 f16 layouts, ISA 7.12.2) ----
// A 16x32 row-major: lane<16 -> K{0..7,16..23}, lane>=16 -> K{8..15,24..31}
__device__ __forceinline__ v16h load_frag_a(const half_t* p, int ld, int lane) {
  const half_t* q = p + (size_t)(lane & 15) * ld + ((lane >> 4) << 3);
  HV x; x.u.lo = *(const v4u*)q; x.u.hi = *(const v4u*)(q + 16);
  return x.h;
}
// B 32x16 loaded from B^T storage (row n contiguous over k):
// lane&15 = column n, lanes 0-15 K{0..7,16..23}, lanes 16-31 K{8..15,24..31}
__device__ __forceinline__ v16h load_frag_bt(const half_t* p, int ld, int lane) {
  const half_t* q = p + (size_t)(lane & 15) * ld + ((lane >> 4) << 3);
  HV x; x.u.lo = *(const v4u*)q; x.u.hi = *(const v4u*)(q + 16);
  return x.h;
}

// ---- block GEMM core: 256 thr, 8 waves (2 x 4), tile 128M x 128N, K step 32 ----
// A row-major (f16, ld=K), Bt = B^T row-major (f16, ld=K). acc: wave tile 64x32.
__device__ __forceinline__ void gemm_block(const half_t* __restrict__ A,
                                           const half_t* __restrict__ Bt,
                                           int K, size_t mBlock, int nBlock,
                                           half_t* sA, v8f acc[8]) {
  const int tid  = threadIdx.x;
  const int lane = tid & 31, wave = tid >> 5;
  const int waveM = wave & 1, waveN = wave >> 1;
#pragma unroll
  for (int i = 0; i < 8; ++i)
#pragma unroll
    for (int e = 0; e < 8; ++e) acc[i][e] = 0.0f;

  const half_t* bBase = Bt + (size_t)(nBlock + waveN * 32) * K;
  const int r = tid >> 1, cOff = (tid & 1) << 4;
  const half_t* aRow = A + (mBlock + (size_t)r) * K + cOff;
  half_t* sRow = sA + r * 32 + cOff;

  for (int k0 = 0; k0 < K; k0 += 32) {
    *(v4u*)sRow       = *(const v4u*)(aRow + k0);
    *(v4u*)(sRow + 8) = *(const v4u*)(aRow + k0 + 8);
    __builtin_prefetch(aRow + k0 + 32, 0, 1);
    __syncthreads();
    v16h bf0 = load_frag_bt(bBase + k0, K, lane);
    v16h bf1 = load_frag_bt(bBase + (size_t)16 * K + k0, K, lane);
    const half_t* aTile = sA + waveM * 64 * 32;
#pragma unroll
    for (int mi = 0; mi < 4; ++mi) {
      v16h af = load_frag_a(aTile + mi * 16 * 32, 32, lane);
      acc[mi * 2 + 0] = __builtin_amdgcn_wmma_f32_16x16x32_f16(
          false, af, false, bf0, (short)0, acc[mi * 2 + 0], false, false);
      acc[mi * 2 + 1] = __builtin_amdgcn_wmma_f32_16x16x32_f16(
          false, af, false, bf1, (short)0, acc[mi * 2 + 1], false, false);
    }
    __syncthreads();
  }
}

// ---- input transpose NCHW -> NHWC ----
__global__ __launch_bounds__(256) void k_nchw2nhwc(const float* __restrict__ x,
                                                   float* __restrict__ xt) {
  __shared__ float t[32][33];
  const int p0 = blockIdx.x * 32, c0 = blockIdx.y * 32, b = blockIdx.z;
  const int tx = threadIdx.x & 31, ty = threadIdx.x >> 5;
  const float* src = x + ((size_t)b * CDIM + c0) * 3136 + p0;
#pragma unroll
  for (int i = 0; i < 4; ++i) {
    int c = ty + i * 8;
    t[c][tx] = src[(size_t)c * 3136 + tx];
  }
  __syncthreads();
  float* dst = xt + ((size_t)b * 3136 + p0) * CDIM + c0;
#pragma unroll
  for (int i = 0; i < 4; ++i) {
    int p = ty + i * 8;
    dst[(size_t)p * CDIM + tx] = t[tx][p];
  }
}

// ---- weight transpose+cast: src R x C (f32) -> dst C x R (f16) ----
__global__ __launch_bounds__(256) void k_wt(const float* __restrict__ src,
                                            half_t* __restrict__ dst, int R, int C) {
  size_t i = (size_t)blockIdx.x * 256 + threadIdx.x;
  if (i >= (size_t)R * C) return;
  int r = (int)(i / C), c = (int)(i % C);
  dst[(size_t)c * R + r] = (half_t)src[i];
}

// ---- LN1: window partition of rolled input, layernorm, cast f16 ----
__global__ __launch_bounds__(256) void k_ln1(const float* __restrict__ XT,
                                             const float* __restrict__ g,
                                             const float* __restrict__ bta,
                                             half_t* __restrict__ XW) {
  const int wave = threadIdx.x >> 5, lane = threadIdx.x & 31;
  const size_t token = (size_t)blockIdx.x * 8 + wave;
  const int wid = (int)(token / 49), n = (int)(token % 49);
  const int b = wid >> 6, wim = wid & 63;
  const int sh = (wim >> 3) * 7 + n / 7, sw = (wim & 7) * 7 + n % 7;
  const int oh = (sh + 3) % 56, ow = (sw + 3) % 56;
  const float* src = XT + ((size_t)b * 3136 + oh * 56 + ow) * CDIM;
  float vals[12], s = 0.f, s2 = 0.f;
#pragma unroll
  for (int i = 0; i < 12; ++i) {
    float v = src[lane + i * 32];
    vals[i] = v; s += v; s2 += v * v;
  }
  for (int off = 16; off; off >>= 1) { s += __shfl_xor(s, off, 32); s2 += __shfl_xor(s2, off, 32); }
  const float mean = s * (1.0f / CDIM);
  const float inv = rsqrtf(s2 * (1.0f / CDIM) - mean * mean + 1e-5f);
  half_t* dst = XW + token * CDIM;
#pragma unroll
  for (int i = 0; i < 12; ++i) {
    int c = lane + i * 32;
    dst[c] = (half_t)((vals[i] - mean) * inv * g[c] + bta[c]);
  }
}

// ---- LN2: plain layernorm over NHWC rows ----
__global__ __launch_bounds__(256) void k_ln2(const float* __restrict__ XR,
                                             const float* __restrict__ g,
                                             const float* __restrict__ bta,
                                             half_t* __restrict__ XN) {
  const int wave = threadIdx.x >> 5, lane = threadIdx.x & 31;
  const size_t token = (size_t)blockIdx.x * 8 + wave;
  const float* src = XR + token * CDIM;
  float vals[12], s = 0.f, s2 = 0.f;
#pragma unroll
  for (int i = 0; i < 12; ++i) {
    float v = src[lane + i * 32];
    vals[i] = v; s += v; s2 += v * v;
  }
  for (int off = 16; off; off >>= 1) { s += __shfl_xor(s, off, 32); s2 += __shfl_xor(s2, off, 32); }
  const float mean = s * (1.0f / CDIM);
  const float inv = rsqrtf(s2 * (1.0f / CDIM) - mean * mean + 1e-5f);
  half_t* dst = XN + token * CDIM;
#pragma unroll
  for (int i = 0; i < 12; ++i) {
    int c = lane + i * 32;
    dst[c] = (half_t)((vals[i] - mean) * inv * g[c] + bta[c]);
  }
}

// ---- QKV GEMM + scatter into padded per-head Q / K / V^T ----
__global__ __launch_bounds__(256) void k_qkv(const half_t* __restrict__ XW,
                                             const half_t* __restrict__ Wt,
                                             const float* __restrict__ bias,
                                             half_t* __restrict__ Q,
                                             half_t* __restrict__ Km,
                                             half_t* __restrict__ VT) {
  __shared__ __align__(16) half_t sA[128 * 32];
  v8f acc[8];
  const size_t mBlock = (size_t)blockIdx.x * 128;
  const int nBlock = blockIdx.y * 128;
  gemm_block(XW, Wt, CDIM, mBlock, nBlock, sA, acc);
  const int lane = threadIdx.x & 31, wave = threadIdx.x >> 5;
  const int waveM = wave & 1, waveN = wave >> 1;
#pragma unroll
  for (int mi = 0; mi < 4; ++mi)
#pragma unroll
    for (int ni = 0; ni < 2; ++ni)
#pragma unroll
      for (int e = 0; e < 8; ++e) {
        size_t m = mBlock + waveM * 64 + mi * 16 + 8 * (lane >> 4) + e;
        int j = nBlock + waveN * 32 + ni * 16 + (lane & 15);
        float v = acc[mi * 2 + ni][e] + bias[j];
        int wid = (int)(m / 49), n = (int)(m % 49);
        int which = j / CDIM, hd = j % CDIM, head = hd >> 5, d = hd & 31;
        size_t base = (size_t)wid * NHEADS + head;
        if (which == 0)      Q[(base * NPAD + n) * HD + d]  = (half_t)(v * SCALE_Q);
        else if (which == 1) Km[(base * NPAD + n) * HD + d] = (half_t)v;
        else                 VT[(base * HD + d) * NPAD + n] = (half_t)v;
      }
}

__device__ __forceinline__ int reg7(int g) { return g < 49 ? 0 : (g < 53 ? 1 : 2); }

// ---- windowed attention: one block per (window, head), 4 waves ----
__global__ __launch_bounds__(128) void k_attn(const half_t* __restrict__ Q,
                                              const half_t* __restrict__ Km,
                                              const half_t* __restrict__ VT,
                                              const float* __restrict__ rpb,
                                              half_t* __restrict__ AOUT) {
  const int head = blockIdx.x, win = blockIdx.y;
  __shared__ __align__(16) float  sS[64 * 64];
  __shared__ __align__(16) half_t sP[64 * 64];
  const int tid = threadIdx.x, lane = tid & 31, wave = tid >> 5;
  const size_t qb = ((size_t)win * NHEADS + head) * (NPAD * HD);
  const size_t vb = ((size_t)win * NHEADS + head) * (HD * NPAD);
  const int m0 = wave * 16;

  // S = q @ k^T  (single K=32 WMMA step per tile)
  v16h af = load_frag_a(Q + qb + (size_t)m0 * HD, HD, lane);
#pragma unroll
  for (int nb = 0; nb < 4; ++nb) {
    v16h bf = load_frag_bt(Km + qb + (size_t)nb * 16 * HD, HD, lane);
    v8f c;
#pragma unroll
    for (int e = 0; e < 8; ++e) c[e] = 0.0f;
    c = __builtin_amdgcn_wmma_f32_16x16x32_f16(false, af, false, bf, (short)0, c, false, false);
#pragma unroll
    for (int e = 0; e < 8; ++e)
      sS[(m0 + 8 * (lane >> 4) + e) * 64 + nb * 16 + (lane & 15)] = c[e];
  }
  __syncthreads();

  // softmax rows with fused relative-position bias + shift mask
  if (tid < 64) {
    const int i = tid;
    half_t* prow = sP + i * 64;
    if (i < 49) {
      const int wim = win & 63, wh = wim >> 3, ww = wim & 7;
      const int ih = i / 7, iw = i % 7;
      const int ri = reg7(wh * 7 + ih) * 3 + reg7(ww * 7 + iw);
      float* row = sS + i * 64;
      float mx = -1e30f;
      for (int j = 0; j < 49; ++j) {
        int jh = j / 7, jw = j % 7;
        int rj = reg7(wh * 7 + jh) * 3 + reg7(ww * 7 + jw);
        float bias = rpb[((ih - jh + 6) * 13 + (iw - jw + 6)) * NHEADS + head];
        float s = row[j] + bias + ((ri != rj) ? -100.0f : 0.0f);
        row[j] = s;
        mx = fmaxf(mx, s);
      }
      float sum = 0.0f;
      for (int j = 0; j < 49; ++j) {
        float e = expf(row[j] - mx);
        row[j] = e; sum += e;
      }
      float inv = 1.0f / sum;
      for (int j = 0; j < 49; ++j) prow[j] = (half_t)(row[j] * inv);
      for (int j = 49; j < 64; ++j) prow[j] = (half_t)0.0f;
    } else {
      for (int j = 0; j < 64; ++j) prow[j] = (half_t)0.0f;
    }
  }
  __syncthreads();

  // O = P @ V  (V stored transposed: HD x NPAD, contiguous B-frags)
  v8f o0, o1;
#pragma unroll
  for (int e = 0; e < 8; ++e) { o0[e] = 0.0f; o1[e] = 0.0f; }
#pragma unroll
  for (int k0 = 0; k0 < 64; k0 += 32) {
    v16h pf = load_frag_a(sP + m0 * 64 + k0, 64, lane);
    v16h b0 = load_frag_bt(VT + vb + k0, NPAD, lane);
    v16h b1 = load_frag_bt(VT + vb + (size_t)16 * NPAD + k0, NPAD, lane);
    o0 = __builtin_amdgcn_wmma_f32_16x16x32_f16(false, pf, false, b0, (short)0, o0, false, false);
    o1 = __builtin_amdgcn_wmma_f32_16x16x32_f16(false, pf, false, b1, (short)0, o1, false, false);
  }
#pragma unroll
  for (int e = 0; e < 8; ++e) {
    int m = m0 + 8 * (lane >> 4) + e;
    if (m < 49) {
      size_t ob = ((size_t)win * 49 + m) * CDIM + head * HD;
      AOUT[ob + (lane & 15)]      = (half_t)o0[e];
      AOUT[ob + 16 + (lane & 15)] = (half_t)o1[e];
    }
  }
}

// ---- proj GEMM + window-reverse + roll-back + shortcut add -> residual ----
__global__ __launch_bounds__(256) void k_proj(const half_t* __restrict__ AOUT,
                                              const half_t* __restrict__ Wt,
                                              const float* __restrict__ bias,
                                              const float* __restrict__ XT,
                                              float* __restrict__ XRES) {
  __shared__ __align__(16) half_t sA[128 * 32];
  v8f acc[8];
  const size_t mBlock = (size_t)blockIdx.x * 128;
  const int nBlock = blockIdx.y * 128;
  gemm_block(AOUT, Wt, CDIM, mBlock, nBlock, sA, acc);
  const int lane = threadIdx.x & 31, wave = threadIdx.x >> 5;
  const int waveM = wave & 1, waveN = wave >> 1;
#pragma unroll
  for (int mi = 0; mi < 4; ++mi)
#pragma unroll
    for (int ni = 0; ni < 2; ++ni)
#pragma unroll
      for (int e = 0; e < 8; ++e) {
        size_t m = mBlock + waveM * 64 + mi * 16 + 8 * (lane >> 4) + e;
        int j = nBlock + waveN * 32 + ni * 16 + (lane & 15);
        float v = acc[mi * 2 + ni][e] + bias[j];
        int wid = (int)(m / 49), n = (int)(m % 49);
        int b = wid >> 6, wim = wid & 63;
        int sh = (wim >> 3) * 7 + n / 7, sw = (wim & 7) * 7 + n % 7;
        int oh = (sh + 3) % 56, ow = (sw + 3) % 56;
        size_t idx = ((size_t)b * 3136 + oh * 56 + ow) * CDIM + j;
        XRES[idx] = v + XT[idx];
      }
}

// ---- fc1 GEMM + bias + exact GELU -> f16 hidden ----
__global__ __launch_bounds__(256) void k_fc1(const half_t* __restrict__ XN,
                                             const half_t* __restrict__ Wt,
                                             const float* __restrict__ bias,
                                             half_t* __restrict__ HIDb) {
  __shared__ __align__(16) half_t sA[128 * 32];
  v8f acc[8];
  const size_t mBlock = (size_t)blockIdx.x * 128;
  const int nBlock = blockIdx.y * 128;
  gemm_block(XN, Wt, CDIM, mBlock, nBlock, sA, acc);
  const int lane = threadIdx.x & 31, wave = threadIdx.x >> 5;
  const int waveM = wave & 1, waveN = wave >> 1;
#pragma unroll
  for (int mi = 0; mi < 4; ++mi)
#pragma unroll
    for (int ni = 0; ni < 2; ++ni)
#pragma unroll
      for (int e = 0; e < 8; ++e) {
        size_t m = mBlock + waveM * 64 + mi * 16 + 8 * (lane >> 4) + e;
        int j = nBlock + waveN * 32 + ni * 16 + (lane & 15);
        float v = acc[mi * 2 + ni][e] + bias[j];
        float g = 0.5f * v * (1.0f + erff(v * 0.70710678118654752f));
        HIDb[m * HIDDIM + j] = (half_t)g;
      }
}

// ---- fc2 GEMM + bias -> fp32 NHWC mlp out ----
__global__ __launch_bounds__(256) void k_fc2(const half_t* __restrict__ HIDb,
                                             const half_t* __restrict__ Wt,
                                             const float* __restrict__ bias,
                                             float* __restrict__ MLPb) {
  __shared__ __align__(16) half_t sA[128 * 32];
  v8f acc[8];
  const size_t mBlock = (size_t)blockIdx.x * 128;
  const int nBlock = blockIdx.y * 128;
  gemm_block(HIDb, Wt, HIDDIM, mBlock, nBlock, sA, acc);
  const int lane = threadIdx.x & 31, wave = threadIdx.x >> 5;
  const int waveM = wave & 1, waveN = wave >> 1;
#pragma unroll
  for (int mi = 0; mi < 4; ++mi)
#pragma unroll
    for (int ni = 0; ni < 2; ++ni)
#pragma unroll
      for (int e = 0; e < 8; ++e) {
        size_t m = mBlock + waveM * 64 + mi * 16 + 8 * (lane >> 4) + e;
        int j = nBlock + waveN * 32 + ni * 16 + (lane & 15);
        MLPb[m * CDIM + j] = acc[mi * 2 + ni][e] + bias[j];
      }
}

// ---- final: out(NCHW) = residual(NHWC) + mlp(NHWC), tiled transpose-add ----
__global__ __launch_bounds__(256) void k_out_add(const float* __restrict__ XRES,
                                                 const float* __restrict__ MLPb,
                                                 float* __restrict__ out) {
  __shared__ float t[32][33];
  const int p0 = blockIdx.x * 32, c0 = blockIdx.y * 32, b = blockIdx.z;
  const int tx = threadIdx.x & 31, ty = threadIdx.x >> 5;
#pragma unroll
  for (int i = 0; i < 4; ++i) {
    int p = ty + i * 8;
    size_t idx = ((size_t)b * 3136 + p0 + p) * CDIM + c0 + tx;
    t[p][tx] = XRES[idx] + MLPb[idx];
  }
  __syncthreads();
#pragma unroll
  for (int i = 0; i < 4; ++i) {
    int c = ty + i * 8;
    out[((size_t)b * CDIM + c0 + c) * 3136 + p0 + tx] = t[tx][c];
  }
}

extern "C" void kernel_launch(void* const* d_in, const int* in_sizes, int n_in,
                              void* d_out, int out_size, void* d_ws, size_t ws_size,
                              hipStream_t stream) {
  (void)in_sizes; (void)n_in; (void)out_size; (void)ws_size;
  const float* x      = (const float*)d_in[0];
  const float* qkv_w  = (const float*)d_in[1];
  const float* qkv_b  = (const float*)d_in[2];
  const float* proj_w = (const float*)d_in[3];
  const float* proj_b = (const float*)d_in[4];
  const float* rpb    = (const float*)d_in[5];
  const float* n1w    = (const float*)d_in[6];
  const float* n1b    = (const float*)d_in[7];
  const float* n2w    = (const float*)d_in[8];
  const float* n2b    = (const float*)d_in[9];
  const float* fc1_w  = (const float*)d_in[10];
  const float* fc1_b  = (const float*)d_in[11];
  const float* fc2_w  = (const float*)d_in[12];
  const float* fc2_b  = (const float*)d_in[13];
  float* out = (float*)d_out;

  char* ws = (char*)d_ws;
  size_t off = 0;
  auto take = [&](size_t nbytes) -> char* {
    char* p = ws + off;
    off = (off + nbytes + 255) & ~(size_t)255;
    return p;
  };
  half_t* Wqkv = (half_t*)take((size_t)3 * CDIM * CDIM * 2);
  half_t* Wproj = (half_t*)take((size_t)CDIM * CDIM * 2);
  half_t* Wfc1 = (half_t*)take((size_t)HIDDIM * CDIM * 2);
  half_t* Wfc2 = (half_t*)take((size_t)CDIM * HIDDIM * 2);
  float*  XT   = (float*)take((size_t)TOK * CDIM * 4);
  half_t* XW   = (half_t*)take((size_t)TOK * CDIM * 2);
  const size_t qkbytes = (size_t)NWIN * NHEADS * NPAD * HD * 2;
  half_t* Qb  = (half_t*)take(qkbytes);
  half_t* Kb  = (half_t*)take(qkbytes);
  half_t* VTb = (half_t*)take(qkbytes);
  half_t* AOUT = (half_t*)take((size_t)TOK * CDIM * 2);
  float*  XRES = (float*)take((size_t)TOK * CDIM * 4);
  half_t* XN   = (half_t*)take((size_t)TOK * CDIM * 2);
  half_t* HIDb = (half_t*)take((size_t)TOK * HIDDIM * 2);
  float*  MLPb = (float*)take((size_t)TOK * CDIM * 4);

  // 1) input transpose NCHW -> NHWC (shortcut)
  k_nchw2nhwc<<<dim3(98, 12, 32), 256, 0, stream>>>(x, XT);
  // 2) weight transposes (B^T, f16)
  k_wt<<<(3 * CDIM * CDIM + 255) / 256, 256, 0, stream>>>(qkv_w, Wqkv, CDIM, 3 * CDIM);
  k_wt<<<(CDIM * CDIM + 255) / 256, 256, 0, stream>>>(proj_w, Wproj, CDIM, CDIM);
  k_wt<<<(CDIM * HIDDIM + 255) / 256, 256, 0, stream>>>(fc1_w, Wfc1, CDIM, HIDDIM);
  k_wt<<<(HIDDIM * CDIM + 255) / 256, 256, 0, stream>>>(fc2_w, Wfc2, HIDDIM, CDIM);
  // 3) LN1 on shifted windows
  k_ln1<<<TOK / 8, 256, 0, stream>>>(XT, n1w, n1b, XW);
  // 4) zero padded q/k/v^T (pad rows must stay zero every call)
  (void)hipMemsetAsync(Qb, 0, qkbytes, stream);
  (void)hipMemsetAsync(Kb, 0, qkbytes, stream);
  (void)hipMemsetAsync(VTb, 0, qkbytes, stream);
  // 5) qkv GEMM (100352x384 @ 384x1152)
  k_qkv<<<dim3(TOK / 128, 9), 256, 0, stream>>>(XW, Wqkv, qkv_b, Qb, Kb, VTb);
  // 6) attention per (window, head)
  k_attn<<<dim3(NHEADS, NWIN), 128, 0, stream>>>(Qb, Kb, VTb, rpb, AOUT);
  // 7) proj GEMM + unshift + shortcut
  k_proj<<<dim3(TOK / 128, 3), 256, 0, stream>>>(AOUT, Wproj, proj_b, XT, XRES);
  // 8) LN2
  k_ln2<<<TOK / 8, 256, 0, stream>>>(XRES, n2w, n2b, XN);
  // 9) fc1 + GELU
  k_fc1<<<dim3(TOK / 128, 12), 256, 0, stream>>>(XN, Wfc1, fc1_b, HIDb);
  // 10) fc2
  k_fc2<<<dim3(TOK / 128, 3), 256, 0, stream>>>(HIDb, Wfc2, fc2_b, MLPb);
  // 11) final residual add + NHWC -> NCHW
  k_out_add<<<dim3(98, 12, 32), 256, 0, stream>>>(XRES, MLPb, out);
}
